// RecurrentBlock_90658169684213
// MI455X (gfx1250) — matrix operand order repaired and checked by reading, not compiled, not run on hardware
//
#include <hip/hip_runtime.h>
#include <hip/hip_bf16.h>

// Problem constants (match reference)
#define TT 512
#define NB 512
#define II 256
#define HH 256
#define KCHUNKS (II / 32)      // 8
#define GTILES  (3 * HH / 16)  // 48 row-tiles per weight matrix
#define NTPW 2                 // column tiles per wave

typedef __attribute__((ext_vector_type(16))) __bf16 v16bf;
typedef __attribute__((ext_vector_type(8)))  __bf16 v8bf;
typedef __attribute__((ext_vector_type(8)))  float  v8f;

__device__ __forceinline__ float sigmoid_f(float x) {
    return 1.0f / (1.0f + __expf(-x));
}
__device__ __forceinline__ float tanh_f(float x) {
    // numerically safe: e=inf -> 1, e=0 -> -1
    float e = __expf(2.0f * x);
    return 1.0f - 2.0f / (e + 1.0f);
}

// ---------------------------------------------------------------------------
// One-time weight pack: W (3H x 256, row-major f32) -> bf16 B-fragments laid
// out fragment-contiguous: dst[((rowTile*KCHUNKS + kChunk)*32 + lane)*16 ..].
// B-fragment (32x16, K x N): lane j holds output column j = W row
// (rowTile*16 + j); lanes 0-15 K=0..7,16..23, lanes 16-31 K=8..15,24..31 of the chunk.
// ---------------------------------------------------------------------------
__global__ __launch_bounds__(32)
void pack_w_kernel(const float* __restrict__ W, __bf16* __restrict__ dst) {
    const int lane = threadIdx.x;
    const int col  = lane & 15;
    const int koff = (lane & 16) ? 8 : 0;
    const int rt = blockIdx.x;  // row tile (0..GTILES-1)
    const int kc = blockIdx.y;  // K chunk (0..KCHUNKS-1)

    const float* p = W + (size_t)(rt * 16 + col) * II + kc * 32 + koff;
    __bf16* q = dst + (((size_t)rt * KCHUNKS + kc) * 32 + lane) * 16;
#pragma unroll
    for (int i = 0; i < 16; i += 2) {
        float2 f = *(const float2*)(p + ((i & 8) << 1) + (i & 7));
        q[i]     = (__bf16)f.x;
        q[i + 1] = (__bf16)f.y;
    }
}

// One-time h0 (N x H f32) -> row-major bf16 into ping-pong buffer 0.
__global__ __launch_bounds__(256)
void cvt_h0_kernel(const float* __restrict__ h0, __bf16* __restrict__ hb) {
    const int i = blockIdx.x * 256 + threadIdx.x;
    hb[i] = (__bf16)h0[i];
}

// A-fragment (16x32 bf16, M x K) from row-major f32 (x input path).
__device__ __forceinline__ v16bf load_a_f32(const float* __restrict__ src,
                                            int ld, int lane) {
    const int row  = lane & 15;
    const int koff = (lane & 16) ? 8 : 0;
    const float* p = src + row * ld;
    v16bf a;
#pragma unroll
    for (int v = 0; v < 8; ++v) {
        const int kk = ((v & 4) << 2) + koff + ((v & 3) << 1); // (v/4)*16 + koff + (v%4)*2
        float2 f = *(const float2*)(p + kk);
        a[2 * v]     = (__bf16)f.x;
        a[2 * v + 1] = (__bf16)f.y;
    }
    return a;
}

// A-fragment from row-major bf16 (h recurrent path): K runs {koff..koff+7}
// and {koff+16..koff+23} are element-contiguous -> two 16-byte loads, no cvt.
__device__ __forceinline__ v16bf load_a_bf16(const __bf16* __restrict__ src,
                                             int ld, int lane) {
    const int row  = lane & 15;
    const int koff = (lane & 16) ? 8 : 0;
    const __bf16* p = src + row * ld + koff;
    const v8bf lo = *(const v8bf*)(p);
    const v8bf hi = *(const v8bf*)(p + 16);
    v16bf a;
#pragma unroll
    for (int i = 0; i < 8; ++i) { a[i] = lo[i]; a[8 + i] = hi[i]; }
    return a;
}

__global__ __launch_bounds__(32)
void gru_step_kernel(const float* __restrict__ x,      // (T*N, I)
                     const float* __restrict__ done,   // (T*N)
                     const v16bf* __restrict__ BihF,   // packed W_ih fragments
                     const v16bf* __restrict__ BhhF,   // packed W_hh fragments
                     const float* __restrict__ bih,    // (3H)
                     const float* __restrict__ bhh,    // (3H)
                     const __bf16* __restrict__ hb_in, // (N, H) bf16 h_{t-1}
                     __bf16* __restrict__ hb_out,      // (N, H) bf16 h_t
                     float* __restrict__ out,          // (T*N, H) ++ (N, H)
                     int t) {
    const int lane = threadIdx.x;
    const int ntb = blockIdx.x * NTPW;  // first H-column tile of this wave
    const int m0  = blockIdx.y * 16;    // batch-row tile

    const float* xt = x    + (size_t)t * NB * II + (size_t)m0 * II;
    const float* dt = done + (size_t)t * NB + m0;
    const __bf16* hb = hb_in + (size_t)m0 * HH;
    float* ho = out + (size_t)t * NB * HH + (size_t)m0 * HH;

    // Zero-init accumulators (first WMMA uses inline C=0); biases in epilogue.
    v8f ai[3][NTPW] = {};  // x-side, per gate / column tile
    v8f ah[3][NTPW] = {};  // h-side

    // Fragment bases: tile index for gate g at column tile nt is (g*16 + nt).
    const v16bf* bi[3][NTPW];
    const v16bf* bh[3][NTPW];
#pragma unroll
    for (int g = 0; g < 3; ++g)
#pragma unroll
        for (int c = 0; c < NTPW; ++c) {
            bi[g][c] = BihF + ((size_t)(g * 16 + ntb + c) * KCHUNKS) * 32 + lane;
            bh[g][c] = BhhF + ((size_t)(g * 16 + ntb + c) * KCHUNKS) * 32 + lane;
        }

#pragma unroll
    for (int kc = 0; kc < KCHUNKS; ++kc) {
        const v16bf Ax = load_a_f32(xt + kc * 32, II, lane);
        const v16bf Ah = load_a_bf16(hb + kc * 32, HH, lane);
        const size_t fo = (size_t)kc * 32;

#pragma unroll
        for (int g = 0; g < 3; ++g)
#pragma unroll
            for (int c = 0; c < NTPW; ++c)
                ai[g][c] = __builtin_amdgcn_wmma_f32_16x16x32_bf16(
                    false, Ax, false, bi[g][c][fo], (short)0, ai[g][c], false, false);
#pragma unroll
        for (int g = 0; g < 3; ++g)
#pragma unroll
            for (int c = 0; c < NTPW; ++c)
                ah[g][c] = __builtin_amdgcn_wmma_f32_16x16x32_bf16(
                    false, Ah, false, bh[g][c][fo], (short)0, ah[g][c], false, false);
    }

    // Epilogue. C/D layout: VGPR e -> M = e (+8 for lanes 16-31); all 8
    // elements of a lane share one output column -> per-lane scalar biases.
    // Done-reset applied here exactly: masked row => h@W terms scaled to 0,
    // biases survive (matching gh = 0@W + b_hh in the reference).
    const int rsel = (lane & 16) ? 8 : 0;
#pragma unroll
    for (int c = 0; c < NTPW; ++c) {
        const int n0  = (ntb + c) * 16;
        const int col = n0 + (lane & 15);
        const float brc = bih[col] + bhh[col];                    // r: both biases
        const float bzc = bih[HH + col] + bhh[HH + col];          // z: both biases
        const float bn  = bih[2 * HH + col];                      // n: input bias
        const float cn  = bhh[2 * HH + col];                      // n: hidden bias
#pragma unroll
        for (int e = 0; e < 8; ++e) {
            const int row = e + rsel;
            const float s  = 1.0f - dt[row];
            const float r  = sigmoid_f(ai[0][c][e] + brc + s * ah[0][c][e]);
            const float z  = sigmoid_f(ai[1][c][e] + bzc + s * ah[1][c][e]);
            const float nn = tanh_f(ai[2][c][e] + bn + r * (s * ah[2][c][e] + cn));
            const float hv = s * (float)hb[row * HH + col];
            const float hn = (1.0f - z) * nn + z * hv;
            ho[row * HH + col] = hn;
            hb_out[(size_t)(m0 + row) * HH + col] = (__bf16)hn;
        }
    }
}

extern "C" void kernel_launch(void* const* d_in, const int* in_sizes, int n_in,
                              void* d_out, int out_size, void* d_ws, size_t ws_size,
                              hipStream_t stream) {
    (void)in_sizes; (void)n_in; (void)out_size; (void)ws_size;

    const float* x    = (const float*)d_in[0];  // inputs       (T*N, I)
    const float* h0   = (const float*)d_in[1];  // hidden_state (N, H)
    const float* done = (const float*)d_in[2];  // done         (T*N,)
    const float* Wih  = (const float*)d_in[3];  // W_ih         (3H, I)
    const float* Whh  = (const float*)d_in[4];  // W_hh         (3H, H)
    const float* bih  = (const float*)d_in[5];  // b_ih         (3H,)
    const float* bhh  = (const float*)d_in[6];  // b_hh         (3H,)
    float* out = (float*)d_out;                 // (T*N, H) ++ (N, H)

    // Workspace: packed bf16 weight fragments (786 KB) + h ping-pong (512 KB).
    const size_t frag_elems = (size_t)GTILES * KCHUNKS * 32 * 16;  // per matrix
    __bf16* BihP = (__bf16*)d_ws;
    __bf16* BhhP = BihP + frag_elems;
    __bf16* hbuf0 = BhhP + frag_elems;
    __bf16* hbuf1 = hbuf0 + (size_t)NB * HH;

    const dim3 pgrid(GTILES, KCHUNKS);
    pack_w_kernel<<<pgrid, 32, 0, stream>>>(Wih, BihP);
    pack_w_kernel<<<pgrid, 32, 0, stream>>>(Whh, BhhP);
    cvt_h0_kernel<<<(NB * HH) / 256, 256, 0, stream>>>(h0, hbuf0);

    const dim3 grid(HH / (16 * NTPW), NB / 16);  // 8 x 32 waves
    for (int t = 0; t < TT; ++t) {
        __bf16* hin  = (t & 1) ? hbuf1 : hbuf0;
        __bf16* hout = (t & 1) ? hbuf0 : hbuf1;
        gru_step_kernel<<<grid, 32, 0, stream>>>(x, done,
                                                 (const v16bf*)BihP, (const v16bf*)BhhP,
                                                 bih, bhh, hin, hout, out, t);
    }

    // h_final = last step's hidden state, appended after the (T*N, H) outputs.
    hipMemcpyAsync(out + (size_t)TT * NB * HH,
                   out + (size_t)(TT - 1) * NB * HH,
                   (size_t)NB * HH * sizeof(float),
                   hipMemcpyDeviceToDevice, stream);
}
